// MultiHeadSessionAttention_47210280517677
// MI455X (gfx1250) — matrix operand order repaired and checked by reading, not compiled, not run on hardware
//
#include <hip/hip_runtime.h>
#include <hip/hip_bf16.h>

// ---------------------------------------------------------------------------
// MultiHeadSessionAttention for MI455X (gfx1250, wave32, WMMA bf16).
//
// Math restructuring (q_len == 1):
//   context = attn @ (A @ Wv)  ==  (attn @ A) @ Wv      (A = node rows)
// so the V projection GEMM (215 GFLOP) shrinks to ~6 GFLOP, and since
// softmax rows sum to 1 the V bias passes straight through (+bv).
// K is never materialized: the K-proj GEMM owns full 64-col head slices
// per tile, so complete q.k dots are computed in its epilogue.
//
// Pipeline (all on `stream`, serialized by stream order):
//   1) prep_weights : Wq/Wk/Wv/Wo fp32 -> bf16, pre-swizzled B-frag layout
//   2) gemm_q       : q = last_hidden @ Wq + bq            (fp32 -> ws)
//   3) gemm_k_scores: scores[b,h,n] = q . (node@Wk + bk)   (fused epilogue)
//   4) attn_p       : softmax -> P = attn @ node           (bf16 -> ws)
//   5) gemm_ctx     : ctx[:,h*64:+64] = P_h @ Wv_h + bv    (bf16 -> ws)
//   6) gemm_o_ln    : out = LN(ctx @ Wo + bo + last_hidden)
//
// Workspace layout (needs ~98 MiB):
//   [0,8M)    : Wq|Wk|Wv|Wo swizzled bf16 (1M elems each)
//   [8M,16M)  : q fp32 [2048*1024]
//   [16M,24M) : scores fp32 [2048*16*50]
//   [24M,88M) : P bf16 [2048*16*1024]
//   [94M,98M) : ctx bf16 [2048*1024]
// ---------------------------------------------------------------------------

typedef __attribute__((ext_vector_type(16))) __bf16 v16bf;
typedef __attribute__((ext_vector_type(8)))  float  v8f;

union FragU { uint4 q[2]; v16bf v; };

__device__ __forceinline__ v8f wmma_bf16(v16bf a, v16bf b, v8f c) {
  return __builtin_amdgcn_wmma_f32_16x16x32_bf16(
      /*neg_a=*/false, a, /*neg_b=*/false, b,
      /*c_mod=*/(short)0, c, /*reuse_a=*/false, /*reuse_b=*/false);
}

__device__ __forceinline__ v8f zero8() {
  v8f z; 
#pragma unroll
  for (int i = 0; i < 8; ++i) z[i] = 0.0f;
  return z;
}

// B fragment: pre-swizzled in ws so lane L reads its 16 bf16 at frag+L*16.
// Stored mapping: element e of lane L  <->  W[K][N],
//   K = ks*32 + (L/16)*16 + e,  N = nb*16 + (L%16)   (contiguous k order, same as the A frags)
__device__ __forceinline__ v16bf load_bfrag(const __bf16* __restrict__ wsz,
                                            int nb, int ks, int lane) {
  const __bf16* p = wsz + ((size_t)(nb * 32 + ks) * 512) + lane * 16;
  FragU f;
  f.q[0] = *(const uint4*)p;
  f.q[1] = *(const uint4*)(p + 8);
  return f.v;
}

// A fragment from row-major bf16 in global: lane L = row (lane%16),
// K chunk c0..c0+15, c0 = ks*32 + (lane>=16 ? 16 : 0).
__device__ __forceinline__ v16bf load_afrag_g(const __bf16* __restrict__ rowptr,
                                              int ks, int lane) {
  const int c0 = ks * 32 + ((lane & 16) ? 16 : 0);
  FragU f;
  f.q[0] = *(const uint4*)(rowptr + c0);
  f.q[1] = *(const uint4*)(rowptr + c0 + 8);
  return f.v;
}

// A fragment from LDS tile [nrows][32] bf16 (current k-step already staged).
__device__ __forceinline__ v16bf load_afrag_lds(const __bf16* tile, int mb, int lane) {
  const __bf16* p = tile + (mb * 16 + (lane & 15)) * 32 + ((lane & 16) ? 16 : 0);
  FragU f;
  f.q[0] = *(const uint4*)p;
  f.q[1] = *(const uint4*)(p + 8);
  return f.v;
}

// ---------------------------------------------------------------------------
// 1) Weight prep: fp32 [K=1024][N=1024] -> bf16 swizzled B-frag layout.
//    4 weights, 1M elements each, laid out contiguously in ws.
// ---------------------------------------------------------------------------
__launch_bounds__(256)
__global__ void prep_weights_kernel(const float* __restrict__ Wq,
                                    const float* __restrict__ Wk,
                                    const float* __restrict__ Wv,
                                    const float* __restrict__ Wo,
                                    __bf16* __restrict__ dst) {
  const unsigned idx  = blockIdx.x * 256u + threadIdx.x;   // < 4M
  const unsigned wsel = idx >> 20;
  const unsigned o    = idx & 0xFFFFFu;
  const float* src = (wsel == 0) ? Wq : (wsel == 1) ? Wk : (wsel == 2) ? Wv : Wo;
  const unsigned e    = o & 15u;
  const unsigned lane = (o >> 4) & 31u;
  const unsigned frag = o >> 9;
  const unsigned nb   = frag >> 5;
  const unsigned ks   = frag & 31u;
  const unsigned K    = ks * 32u + (lane >> 4) * 16u + e;
  const unsigned N    = nb * 16u + (lane & 15u);
  dst[idx] = (__bf16)src[(size_t)K * 1024u + N];
}

// ---------------------------------------------------------------------------
// Shared 128x128-tile mainloop for fp32-A GEMMs (Q-proj and K-proj).
// 8 waves: 4 (M) x 2 (N); wave tile 32x64 -> acc[2][4].
// A tile (128x32 fp32) staged to LDS as bf16 row-major per k-step.
// ---------------------------------------------------------------------------
__device__ __forceinline__ void gemm128_mainloop(const float* __restrict__ A,
                                                 const __bf16* __restrict__ Bswz,
                                                 int panel, int nchunk,
                                                 __bf16* Atile, v8f (&acc)[2][4]) {
  const int tid  = threadIdx.x;
  const int lane = tid & 31;
  const int w    = tid >> 5;
  const int wm   = w >> 1;   // 0..3
  const int wn   = w & 1;    // 0..1

  for (int ks = 0; ks < 32; ++ks) {
    __syncthreads();
    {  // cooperative fp32->bf16 staging: thread -> (row = tid/2, half = tid&1)
      const int r = tid >> 1, half = tid & 1;
      const float4* s4 =
          (const float4*)(A + (size_t)(panel * 128 + r) * 1024 + ks * 32 + half * 16);
      if (ks < 31) __builtin_prefetch((const void*)(s4 + 8), 0, 0);
      union { __bf16 h[16]; uint4 u[2]; } pk;
#pragma unroll
      for (int c = 0; c < 4; ++c) {
        float4 f = s4[c];
        pk.h[c * 4 + 0] = (__bf16)f.x;
        pk.h[c * 4 + 1] = (__bf16)f.y;
        pk.h[c * 4 + 2] = (__bf16)f.z;
        pk.h[c * 4 + 3] = (__bf16)f.w;
      }
      *(uint4*)&Atile[r * 32 + half * 16]     = pk.u[0];
      *(uint4*)&Atile[r * 32 + half * 16 + 8] = pk.u[1];
    }
    __syncthreads();
    const v16bf af0 = load_afrag_lds(Atile, wm * 2 + 0, lane);
    const v16bf af1 = load_afrag_lds(Atile, wm * 2 + 1, lane);
#pragma unroll
    for (int j = 0; j < 4; ++j) {
      const v16bf bf = load_bfrag(Bswz, nchunk * 8 + wn * 4 + j, ks, lane);
      acc[0][j] = wmma_bf16(af0, bf, acc[0][j]);
      acc[1][j] = wmma_bf16(af1, bf, acc[1][j]);
    }
  }
}

// ---------------------------------------------------------------------------
// 2) Q projection: q = last_hidden @ Wq + bq  (M=2048, grid: 8 x 16)
// ---------------------------------------------------------------------------
__launch_bounds__(256)
__global__ void gemm_q_kernel(const float* __restrict__ A,
                              const __bf16* __restrict__ Bswz,
                              const float* __restrict__ bias,
                              float* __restrict__ D) {
  __shared__ __bf16 Atile[128 * 32];
  const int nchunk = blockIdx.x, panel = blockIdx.y;
  const int lane = threadIdx.x & 31, w = threadIdx.x >> 5;
  const int wm = w >> 1, wn = w & 1;
  v8f acc[2][4];
#pragma unroll
  for (int m = 0; m < 2; ++m)
#pragma unroll
    for (int j = 0; j < 4; ++j) acc[m][j] = zero8();

  gemm128_mainloop(A, Bswz, panel, nchunk, Atile, acc);

  const int rofs = (lane & 16) ? 8 : 0;
#pragma unroll
  for (int m = 0; m < 2; ++m)
#pragma unroll
    for (int j = 0; j < 4; ++j)
#pragma unroll
      for (int r = 0; r < 8; ++r) {
        const int row = panel * 128 + wm * 32 + m * 16 + r + rofs;
        const int col = nchunk * 128 + wn * 64 + j * 16 + (lane & 15);
        D[(size_t)row * 1024 + col] = acc[m][j][r] + bias[col];
      }
}

// ---------------------------------------------------------------------------
// 3) K projection with fused scores. M=102400 (grid: 8 x 800).
//    Each 128-col chunk covers heads 2*nchunk, 2*nchunk+1 fully, so the
//    complete q.k dot for those heads is computed here; K is never stored.
// ---------------------------------------------------------------------------
__launch_bounds__(256)
__global__ void gemm_k_scores_kernel(const float* __restrict__ A,
                                     const __bf16* __restrict__ Bswz,
                                     const float* __restrict__ bias,
                                     const float* __restrict__ qbuf,
                                     float* __restrict__ scores) {
  __shared__ __bf16 Atile[128 * 32];
  __shared__ __bf16 Kt[128 * 128];  // K tile (with bias), bf16
  const int nchunk = blockIdx.x, panel = blockIdx.y;
  const int tid = threadIdx.x;
  const int lane = tid & 31, w = tid >> 5;
  const int wm = w >> 1, wn = w & 1;
  v8f acc[2][4];
#pragma unroll
  for (int m = 0; m < 2; ++m)
#pragma unroll
    for (int j = 0; j < 4; ++j) acc[m][j] = zero8();

  gemm128_mainloop(A, Bswz, panel, nchunk, Atile, acc);

  const int rofs = (lane & 16) ? 8 : 0;
#pragma unroll
  for (int m = 0; m < 2; ++m)
#pragma unroll
    for (int j = 0; j < 4; ++j)
#pragma unroll
      for (int r = 0; r < 8; ++r) {
        const int lrow = wm * 32 + m * 16 + r + rofs;
        const int lcol = wn * 64 + j * 16 + (lane & 15);
        Kt[lrow * 128 + lcol] = (__bf16)(acc[m][j][r] + bias[nchunk * 128 + lcol]);
      }
  __syncthreads();

  // 256 threads -> 128 rows x 2 local heads; full 64-wide dot with q.
  const int r2 = tid >> 1, hl = tid & 1;
  const int grow = panel * 128 + r2;          // global node row = b*50 + n
  const int b = grow / 50, n = grow - b * 50;
  const int h = nchunk * 2 + hl;
  const float* qrow = qbuf + (size_t)b * 1024 + h * 64;
  float dot = 0.0f;
#pragma unroll
  for (int d = 0; d < 64; ++d) dot += qrow[d] * (float)Kt[r2 * 128 + hl * 64 + d];
  scores[((size_t)b * 16 + h) * 50 + n] = dot;
}

// ---------------------------------------------------------------------------
// 4) Softmax + P = attn @ node_hidden[b]    (one workgroup per batch b)
// ---------------------------------------------------------------------------
__launch_bounds__(256)
__global__ void attn_p_kernel(const float* __restrict__ scores,
                              const int* __restrict__ mask,
                              const float* __restrict__ temp,
                              const float* __restrict__ node,
                              __bf16* __restrict__ P) {
  const int b = blockIdx.x;
  const int tid = threadIdx.x;
  __shared__ float attn[16][50];

  if (tid < 16) {
    const int h = tid;
    const float sc = temp[h] * 8.0f + 1e-8f;  // temperature*sqrt(64) + eps
    const float* srow = scores + ((size_t)b * 16 + h) * 50;
    const int* mrow = mask + b * 50;
    float mx = -3.0e38f;
    for (int n = 0; n < 50; ++n) {
      float v = srow[n] / sc;
      if (mrow[n] == 0) v = -1e9f;
      mx = fmaxf(mx, v);
    }
    float sum = 0.0f;
    for (int n = 0; n < 50; ++n) {
      float v = srow[n] / sc;
      if (mrow[n] == 0) v = -1e9f;
      sum += __expf(v - mx);
    }
    const float inv = 1.0f / sum;
    for (int n = 0; n < 50; ++n) {
      float v = srow[n] / sc;
      if (mrow[n] == 0) v = -1e9f;
      attn[h][n] = __expf(v - mx) * inv;
    }
  }
  __syncthreads();

  // P[b,h,c] = sum_n attn[h][n] * node[b,n,c]; each thread owns 4 columns.
#pragma unroll
  for (int cc = 0; cc < 4; ++cc) {
    const int c = tid + cc * 256;
    float pacc[16];
#pragma unroll
    for (int h = 0; h < 16; ++h) pacc[h] = 0.0f;
    for (int n = 0; n < 50; ++n) {
      const float x = node[((size_t)b * 50 + n) * 1024 + c];
#pragma unroll
      for (int h = 0; h < 16; ++h) pacc[h] += attn[h][n] * x;
    }
#pragma unroll
    for (int h = 0; h < 16; ++h)
      P[((size_t)b * 16 + h) * 1024 + c] = (__bf16)pacc[h];
  }
}

// ---------------------------------------------------------------------------
// 5) Per-head context GEMM: ctx[:, h*64:+64] = P_h @ Wv[:, h*64:+64] + bv
//    grid: (panel=16, head=16); 8 waves, each owns 16 rows x 64 cols.
// ---------------------------------------------------------------------------
__launch_bounds__(256)
__global__ void gemm_ctx_kernel(const __bf16* __restrict__ P,
                                const __bf16* __restrict__ Wv_swz,
                                const float* __restrict__ bv,
                                __bf16* __restrict__ ctx) {
  const int panel = blockIdx.x, head = blockIdx.y;
  const int lane = threadIdx.x & 31, w = threadIdx.x >> 5;
  const int brow = panel * 128 + w * 16 + (lane & 15);
  const __bf16* arow = P + ((size_t)brow * 16 + head) * 1024;

  v8f acc[4];
#pragma unroll
  for (int j = 0; j < 4; ++j) acc[j] = zero8();

  for (int ks = 0; ks < 32; ++ks) {
    const v16bf af = load_afrag_g(arow, ks, lane);
#pragma unroll
    for (int j = 0; j < 4; ++j) {
      const v16bf bf = load_bfrag(Wv_swz, head * 4 + j, ks, lane);
      acc[j] = wmma_bf16(af, bf, acc[j]);
    }
  }

  const int rofs = (lane & 16) ? 8 : 0;
#pragma unroll
  for (int j = 0; j < 4; ++j)
#pragma unroll
    for (int r = 0; r < 8; ++r) {
      const int bb = panel * 128 + w * 16 + r + rofs;
      const int col = head * 64 + j * 16 + (lane & 15);
      ctx[(size_t)bb * 1024 + col] = (__bf16)(acc[j][r] + bv[col]);
    }
}

// ---------------------------------------------------------------------------
// 6) Output projection + residual + LayerNorm. Workgroup owns 16 full rows
//    (needed for LN). 8 waves x 128 cols each. grid: 128.
// ---------------------------------------------------------------------------
__launch_bounds__(256)
__global__ void gemm_o_ln_kernel(const __bf16* __restrict__ ctx,
                                 const __bf16* __restrict__ Wo_swz,
                                 const float* __restrict__ bo,
                                 const float* __restrict__ last,
                                 const float* __restrict__ gamma,
                                 const float* __restrict__ beta,
                                 float* __restrict__ out) {
  __shared__ float ob[16 * 1024];
  __shared__ float red[2][16][17];
  __shared__ float mu[16], rs[16];
  const int panel = blockIdx.x;
  const int tid = threadIdx.x;
  const int lane = tid & 31, w = tid >> 5;
  const __bf16* arow = ctx + (size_t)(panel * 16 + (lane & 15)) * 1024;

  v8f acc[8];
#pragma unroll
  for (int j = 0; j < 8; ++j) acc[j] = zero8();

  for (int ks = 0; ks < 32; ++ks) {
    const v16bf af = load_afrag_g(arow, ks, lane);
#pragma unroll
    for (int j = 0; j < 8; ++j) {
      const v16bf bf = load_bfrag(Wo_swz, w * 8 + j, ks, lane);
      acc[j] = wmma_bf16(af, bf, acc[j]);
    }
  }

  const int rofs = (lane & 16) ? 8 : 0;
#pragma unroll
  for (int j = 0; j < 8; ++j)
#pragma unroll
    for (int r = 0; r < 8; ++r) {
      const int row = r + rofs;
      const int col = w * 128 + j * 16 + (lane & 15);
      ob[row * 1024 + col] =
          acc[j][r] + bo[col] + last[(size_t)(panel * 16 + row) * 1024 + col];
    }
  __syncthreads();

  {  // partial sums: 16 segments of 64 per row
    const int row = tid & 15, seg = tid >> 4;
    float s = 0.0f, s2 = 0.0f;
#pragma unroll
    for (int k = 0; k < 64; ++k) {
      const float x = ob[row * 1024 + seg * 64 + k];
      s += x;
      s2 += x * x;
    }
    red[0][row][seg] = s;
    red[1][row][seg] = s2;
  }
  __syncthreads();
  if (tid < 16) {
    float s = 0.0f, s2 = 0.0f;
#pragma unroll
    for (int k = 0; k < 16; ++k) { s += red[0][tid][k]; s2 += red[1][tid][k]; }
    const float m = s * (1.0f / 1024.0f);
    const float var = s2 * (1.0f / 1024.0f) - m * m;
    mu[tid] = m;
    rs[tid] = rsqrtf(var + 1e-5f);
  }
  __syncthreads();
  for (int k = 0; k < 64; ++k) {
    const int e = k * 256 + tid;
    const int row = e >> 10, col = e & 1023;
    out[(size_t)(panel * 16 + row) * 1024 + col] =
        (ob[e] - mu[row]) * rs[row] * gamma[col] + beta[col];
  }
}

// ---------------------------------------------------------------------------
extern "C" void kernel_launch(void* const* d_in, const int* in_sizes, int n_in,
                              void* d_out, int out_size, void* d_ws, size_t ws_size,
                              hipStream_t stream) {
  (void)in_sizes; (void)n_in; (void)out_size; (void)ws_size;
  const float* node  = (const float*)d_in[0];
  const float* last  = (const float*)d_in[1];
  const int*   mask  = (const int*)d_in[2];
  const float* Wq    = (const float*)d_in[3];
  const float* bq    = (const float*)d_in[4];
  const float* Wk    = (const float*)d_in[5];
  const float* bk    = (const float*)d_in[6];
  const float* Wv    = (const float*)d_in[7];
  const float* bv    = (const float*)d_in[8];
  const float* Wo    = (const float*)d_in[9];
  const float* bo    = (const float*)d_in[10];
  const float* temp  = (const float*)d_in[11];
  const float* gamma = (const float*)d_in[12];
  const float* beta  = (const float*)d_in[13];

  char* ws = (char*)d_ws;
  __bf16* wswz   = (__bf16*)ws;                       // 4 x 1M bf16 (8 MiB)
  float*  qbuf   = (float*)(ws + (size_t)(8)  * 1024 * 1024);
  float*  scores = (float*)(ws + (size_t)(16) * 1024 * 1024);
  __bf16* P      = (__bf16*)(ws + (size_t)(24) * 1024 * 1024);
  __bf16* ctx    = (__bf16*)(ws + (size_t)(94) * 1024 * 1024);
  const __bf16* WqS = wswz;
  const __bf16* WkS = wswz + (size_t)1 * 1024 * 1024;
  const __bf16* WvS = wswz + (size_t)2 * 1024 * 1024;
  const __bf16* WoS = wswz + (size_t)3 * 1024 * 1024;

  prep_weights_kernel<<<16384, 256, 0, stream>>>(Wq, Wk, Wv, Wo, wswz);
  gemm_q_kernel<<<dim3(8, 16), 256, 0, stream>>>(last, WqS, bq, qbuf);
  gemm_k_scores_kernel<<<dim3(8, 800), 256, 0, stream>>>(node, WkS, bk, qbuf, scores);
  attn_p_kernel<<<2048, 256, 0, stream>>>(scores, mask, temp, node, P);
  gemm_ctx_kernel<<<dim3(16, 16), 256, 0, stream>>>(P, WvS, bv, ctx);
  gemm_o_ln_kernel<<<128, 256, 0, stream>>>(ctx, WoS, bo, last, gamma, beta,
                                            (float*)d_out);
}
